// MultiHeadAttention_24077586661981
// MI455X (gfx1250) — matrix operand order repaired and checked by reading, not compiled, not run on hardware
//
#include <hip/hip_runtime.h>
#include <hip/hip_bf16.h>

typedef __attribute__((ext_vector_type(16))) _Float16 v16h;
typedef __attribute__((ext_vector_type(8)))  _Float16 v8h;
typedef __attribute__((ext_vector_type(4)))  _Float16 v4h;
typedef __attribute__((ext_vector_type(8)))  float    v8f;

constexpr int Bn = 2;
constexpr int Sn = 2048;
constexpr int Dm = 1024;
constexpr int Hn = 16;
constexpr int Dh = 64;   // head dim

// ---------------------------------------------------------------------------
// CDNA5 async global->LDS (ASYNCcnt-tracked), GVS addressing mode.
// ---------------------------------------------------------------------------
__device__ __forceinline__ void async_b128(unsigned lds, unsigned voff,
                                           const void* base) {
  asm volatile("global_load_async_to_lds_b128 %0, %1, %2"
               :: "v"(lds), "v"(voff), "s"(base) : "memory");
}
__device__ __forceinline__ void wait_async0() {
  asm volatile("s_wait_asynccnt 0x0" ::: "memory");
}
__device__ __forceinline__ void wait_async4() {
  asm volatile("s_wait_asynccnt 0x4" ::: "memory");
}
__device__ __forceinline__ void wait_async6() {
  asm volatile("s_wait_asynccnt 0x6" ::: "memory");
}

// ---------------------------------------------------------------------------
// fp32 -> f16 cast, 4 elements/thread (n divisible by 4 here)
// ---------------------------------------------------------------------------
__global__ void cvt_f32_f16(const float* __restrict__ src,
                            _Float16* __restrict__ dst, int n4) {
  int i = blockIdx.x * blockDim.x + threadIdx.x;
  if (i < n4) {
    float4 f = ((const float4*)src)[i];
    v4h h;
    h[0] = (_Float16)f.x; h[1] = (_Float16)f.y;
    h[2] = (_Float16)f.z; h[3] = (_Float16)f.w;
    ((v4h*)dst)[i] = h;
  }
}

// ---------------------------------------------------------------------------
// WMMA fragment helpers (layouts per CDNA5 ISA 7.12.2, 16-bit A/B, wave32)
// ---------------------------------------------------------------------------
__device__ __forceinline__ v16h make_afrag(const _Float16* row, int lane) {
  const int kb = (lane < 16) ? 0 : 8;
  v8h lo = *(const v8h*)(row + kb);
  v8h hi = *(const v8h*)(row + kb + 16);
  v16h a;
#pragma unroll
  for (int i = 0; i < 8; ++i) { a[i] = lo[i]; a[i + 8] = hi[i]; }
  return a;
}

__device__ __forceinline__ v16h make_bfrag(const _Float16* row, int lane) {
  const int ks = (lane < 16) ? 0 : 8;
  v8h lo = *(const v8h*)(row + ks);
  v8h hi = *(const v8h*)(row + ks + 16);
  v16h b;
#pragma unroll
  for (int i = 0; i < 8; ++i) { b[i] = lo[i]; b[i + 8] = hi[i]; }
  return b;
}

__device__ __forceinline__ v8f wmma_f16(v16h a, v16h b, v8f c) {
  return __builtin_amdgcn_wmma_f32_16x16x32_f16(
      false, a, false, b, (short)0, c, false, false);
}

// ---------------------------------------------------------------------------
// GEMM: C[M,N] = X[M,K] @ W[N,K]^T + bias[N]
// 128 threads (4 waves), block tile 128x64, wave tile 32x64 (2 A-frags share
// each B-frag: 8 WMMA per 12 ds_load_b128). Double-buffered LDS tiles fed by
// async global->LDS, pipelined with s_wait_asynccnt 6.
// ---------------------------------------------------------------------------
template <bool F32OUT>
__global__ __launch_bounds__(128)
void gemm_xwt(const _Float16* __restrict__ X, const _Float16* __restrict__ W,
              const float* __restrict__ bias, void* __restrict__ out,
              int M, int N, int K) {
  __shared__ __align__(16) _Float16 As[2][128][40];
  __shared__ __align__(16) _Float16 Bs[2][64][40];

  const int tid  = threadIdx.x;
  const int wave = tid >> 5;
  const int lane = tid & 31;
  const int mbase = blockIdx.x * 128;
  const int nbase = blockIdx.y * 64;

  // Staging coords: A = 128 rows x 4 chunks (4/thread), B = 64 x 4 (2/thread).
  unsigned ldsA[4], offA[4], ldsB[2], offB[2];
#pragma unroll
  for (int i = 0; i < 4; ++i) {
    int c = tid + 128 * i;           // 0..511
    int r = c >> 2, ch = c & 3;
    ldsA[i] = (unsigned)(uintptr_t)&As[0][r][ch * 8];
    offA[i] = (unsigned)(((size_t)(mbase + r) * K + ch * 8) * 2);
  }
#pragma unroll
  for (int i = 0; i < 2; ++i) {
    int c = tid + 128 * i;           // 0..255
    int r = c >> 2, ch = c & 3;
    ldsB[i] = (unsigned)(uintptr_t)&Bs[0][r][ch * 8];
    offB[i] = (unsigned)(((size_t)(nbase + r) * K + ch * 8) * 2);
  }
  const unsigned strideA = (unsigned)(sizeof(_Float16) * 128 * 40);
  const unsigned strideB = (unsigned)(sizeof(_Float16) * 64 * 40);

  v8f acc[2][4];
#pragma unroll
  for (int g = 0; g < 2; ++g)
#pragma unroll
    for (int nt = 0; nt < 4; ++nt)
#pragma unroll
      for (int i = 0; i < 8; ++i) acc[g][nt][i] = 0.f;

  const int nlan = lane & 15;
  const int mloc0 = wave * 32 + nlan;
  const int nIter = K / 32;

  // Prologue: stage tile 0 into buffer 0 (6 async instructions per thread).
#pragma unroll
  for (int i = 0; i < 4; ++i) async_b128(ldsA[i], offA[i], X);
#pragma unroll
  for (int i = 0; i < 2; ++i) async_b128(ldsB[i], offB[i], W);

  for (int it = 0; it < nIter; ++it) {
    const int cur = it & 1;
    if (it + 1 < nIter) {
      const unsigned kadd = (unsigned)((it + 1) * 32 * 2);
      const unsigned la = (cur ^ 1) * strideA;
      const unsigned lb = (cur ^ 1) * strideB;
#pragma unroll
      for (int i = 0; i < 4; ++i) async_b128(ldsA[i] + la, offA[i] + kadd, X);
#pragma unroll
      for (int i = 0; i < 2; ++i) async_b128(ldsB[i] + lb, offB[i] + kadd, W);
      wait_async6();   // current tile's 6 loads done; next tile in flight
    } else {
      wait_async0();
    }
    __syncthreads();

    v16h a0 = make_afrag(&As[cur][mloc0][0], lane);
    v16h a1 = make_afrag(&As[cur][mloc0 + 16][0], lane);
#pragma unroll
    for (int nt = 0; nt < 4; ++nt) {
      v16h b = make_bfrag(&Bs[cur][nt * 16 + nlan][0], lane);
      acc[0][nt] = wmma_f16(a0, b, acc[0][nt]);
      acc[1][nt] = wmma_f16(a1, b, acc[1][nt]);
    }
    __syncthreads();
  }

  const int moff = (lane < 16) ? 0 : 8;
#pragma unroll
  for (int g = 0; g < 2; ++g)
#pragma unroll
    for (int nt = 0; nt < 4; ++nt) {
      int n = nbase + nt * 16 + nlan;
      float bv = bias[n];
#pragma unroll
      for (int i = 0; i < 8; ++i) {
        int row = mbase + wave * 32 + g * 16 + i + moff;
        float val = acc[g][nt][i] + bv;
        if (F32OUT)
          ((float*)out)[(size_t)row * N + n] = val;
        else
          ((_Float16*)out)[(size_t)row * N + n] = (_Float16)val;
      }
    }
}

// ---------------------------------------------------------------------------
// Causal flash attention, one block = (b, h, 64-row q tile), 128 threads.
// Pipelined: K(j+1) async into double-buffered Ks and V(j+1) into registers
// while tile j is computed; V(j) transposed into LDS at top of iteration j.
// ---------------------------------------------------------------------------
__global__ __launch_bounds__(128)
void attn_kernel(const _Float16* __restrict__ Qg, const _Float16* __restrict__ Kg,
                 const _Float16* __restrict__ Vg, _Float16* __restrict__ Og) {
  __shared__ __align__(16) _Float16 Qs[64][72];
  __shared__ __align__(16) _Float16 Ks[2][64][72];
  __shared__ __align__(16) _Float16 Vt[64][72];   // transposed V: Vt[dh][key]
  __shared__ __align__(16) _Float16 Ps[64][72];   // exp(scores)
  __shared__ float Ss[64][68];
  __shared__ float mrow[64], lrow[64], arow[64];

  const int tid  = threadIdx.x;
  const int wave = tid >> 5;
  const int lane = tid & 31;
  const int nlan = lane & 15;
  const int moff = (lane < 16) ? 0 : 8;

  const int qt = blockIdx.x;
  const int h  = blockIdx.y;
  const int b  = blockIdx.z;
  const int qbase = qt * 64;
  const size_t headoff = (size_t)h * Dh;
  const unsigned kbuf = (unsigned)(sizeof(_Float16) * 64 * 72);

  // Per-thread staging coordinates: 4 chunks (8 halves) per 64x64 tile.
  int rr[4], cc[4];
  unsigned ldsK[4], koff0[4], vidx0[4];
#pragma unroll
  for (int i = 0; i < 4; ++i) {
    int c = tid + 128 * i;
    rr[i] = c >> 3; cc[i] = c & 7;
    ldsK[i]  = (unsigned)(uintptr_t)&Ks[0][rr[i]][cc[i] * 8];
    koff0[i] = (unsigned)(((size_t)(b * Sn + rr[i]) * Dm + headoff + cc[i] * 8) * 2);
    vidx0[i] = (unsigned)((size_t)(b * Sn + rr[i]) * Dm + headoff + cc[i] * 8);
  }

  // Prologue: async Q tile + K(0); V(0) into registers.
  v8h vv[4];
#pragma unroll
  for (int i = 0; i < 4; ++i) {
    unsigned ldsQ = (unsigned)(uintptr_t)&Qs[rr[i]][cc[i] * 8];
    async_b128(ldsQ, koff0[i] + (unsigned)(qbase * Dm * 2), Qg);
    async_b128(ldsK[i], koff0[i], Kg);
    vv[i] = *(const v8h*)(Vg + vidx0[i]);
  }
  if (tid < 64) { mrow[tid] = -3.0e38f; lrow[tid] = 0.f; }

  v8f o[4];
#pragma unroll
  for (int nt = 0; nt < 4; ++nt)
#pragma unroll
    for (int i = 0; i < 8; ++i) o[nt][i] = 0.f;

  wait_async0();
  __syncthreads();

  const int mloc = wave * 16 + nlan;

  for (int j = 0; j <= qt; ++j) {
    const int kbase = j * 64;
    const int cur = j & 1;

    // Transpose-store V(j) (loaded last iteration) into Vt.
#pragma unroll
    for (int i = 0; i < 4; ++i)
#pragma unroll
      for (int e = 0; e < 8; ++e) Vt[cc[i] * 8 + e][rr[i]] = vv[i][e];

    // Prefetch tile j+1: K async into other buffer, V into registers.
    if (j < qt) {
      const unsigned jb = (unsigned)((j + 1) * 64 * Dm * 2);
      const unsigned lb = (cur ^ 1) * kbuf;
      const unsigned jv = (unsigned)((j + 1) * 64 * Dm);
#pragma unroll
      for (int i = 0; i < 4; ++i) async_b128(ldsK[i] + lb, koff0[i] + jb, Kg);
#pragma unroll
      for (int i = 0; i < 4; ++i) vv[i] = *(const v8h*)(Vg + vidx0[i] + jv);
      wait_async4();   // K(j) resident; K(j+1) in flight
    } else {
      wait_async0();
    }
    __syncthreads();

    // scores = Q @ K^T * (1/sqrt(64)), causal mask, -> Ss
#pragma unroll
    for (int nt = 0; nt < 4; ++nt) {
      v8f s;
#pragma unroll
      for (int i = 0; i < 8; ++i) s[i] = 0.f;
#pragma unroll
      for (int kk = 0; kk < 64; kk += 32) {
        v16h a = make_afrag(&Qs[mloc][kk], lane);
        v16h bb = make_bfrag(&Ks[cur][nt * 16 + nlan][kk], lane);
        s = wmma_f16(a, bb, s);
      }
      int kcol = kbase + nt * 16 + nlan;
#pragma unroll
      for (int i = 0; i < 8; ++i) {
        int m = i + moff;
        int qrow = qbase + wave * 16 + m;
        float val = s[i] * 0.125f;
        if (kcol > qrow) val = -3.0e38f;
        Ss[wave * 16 + m][nt * 16 + nlan] = val;
      }
    }
    __syncthreads();

    // Online softmax; lane r<16 of each wave owns one row.
    if (lane < 16) {
      int r = wave * 16 + lane;
      float mo = mrow[r];
      float mx = mo;
#pragma unroll 8
      for (int c = 0; c < 64; ++c) mx = fmaxf(mx, Ss[r][c]);
      float al = __expf(mo - mx);
      float sum = 0.f;
#pragma unroll 8
      for (int c = 0; c < 64; ++c) {
        float e = __expf(Ss[r][c] - mx);
        Ps[r][c] = (_Float16)e;
        sum += e;
      }
      lrow[r] = lrow[r] * al + sum;
      mrow[r] = mx;
      arow[r] = al;
    }
    __syncthreads();

    // Rescale accumulator, then O += P @ V.
    float al8[8];
#pragma unroll
    for (int i = 0; i < 8; ++i) al8[i] = arow[wave * 16 + i + moff];
#pragma unroll
    for (int nt = 0; nt < 4; ++nt)
#pragma unroll
      for (int i = 0; i < 8; ++i) o[nt][i] *= al8[i];

#pragma unroll
    for (int kk = 0; kk < 64; kk += 32) {
      v16h a = make_afrag(&Ps[mloc][kk], lane);
#pragma unroll
      for (int nt = 0; nt < 4; ++nt) {
        v16h bb = make_bfrag(&Vt[nt * 16 + nlan][kk], lane);
        o[nt] = wmma_f16(a, bb, o[nt]);
      }
    }
    __syncthreads();   // before next iteration overwrites Ks/Vt/Ps
  }

  // Normalize by l and store head-concat f16 [B,S,D].
  float li[8];
#pragma unroll
  for (int i = 0; i < 8; ++i) li[i] = 1.f / lrow[wave * 16 + i + moff];
#pragma unroll
  for (int nt = 0; nt < 4; ++nt)
#pragma unroll
    for (int i = 0; i < 8; ++i) {
      int row = qbase + wave * 16 + i + moff;
      Og[(size_t)(b * Sn + row) * Dm + headoff + nt * 16 + nlan] =
          (_Float16)(o[nt][i] * li[i]);
    }
}

// ---------------------------------------------------------------------------
// Host-side orchestration
// ---------------------------------------------------------------------------
extern "C" void kernel_launch(void* const* d_in, const int* in_sizes, int n_in,
                              void* d_out, int out_size, void* d_ws, size_t ws_size,
                              hipStream_t stream) {
  (void)in_sizes; (void)n_in; (void)out_size; (void)ws_size;

  const float* q  = (const float*)d_in[0];
  const float* k  = (const float*)d_in[1];
  const float* v  = (const float*)d_in[2];
  // d_in[3] = attn_mask (causal; implemented analytically)
  const float* Wq = (const float*)d_in[4];
  const float* bq = (const float*)d_in[5];
  const float* Wk = (const float*)d_in[6];
  const float* bk = (const float*)d_in[7];
  const float* Wv = (const float*)d_in[8];
  const float* bv = (const float*)d_in[9];
  const float* Wo = (const float*)d_in[10];
  const float* bo = (const float*)d_in[11];

  const int NX = Bn * Sn * Dm;
  const int NW = Dm * Dm;
  char* ws = (char*)d_ws;
  const size_t MB = 1u << 20;
  _Float16* hq  = (_Float16*)(ws + 0 * MB);
  _Float16* hk  = (_Float16*)(ws + 8 * MB);
  _Float16* hv  = (_Float16*)(ws + 16 * MB);
  _Float16* hWq = (_Float16*)(ws + 24 * MB);
  _Float16* hWk = (_Float16*)(ws + 26 * MB);
  _Float16* hWv = (_Float16*)(ws + 28 * MB);
  _Float16* hWo = (_Float16*)(ws + 30 * MB);
  _Float16* hQ  = (_Float16*)(ws + 32 * MB);
  _Float16* hK  = (_Float16*)(ws + 40 * MB);
  _Float16* hV  = (_Float16*)(ws + 48 * MB);
  _Float16* hA  = (_Float16*)(ws + 56 * MB);

  cvt_f32_f16<<<(NX / 4 + 255) / 256, 256, 0, stream>>>(q, hq, NX / 4);
  cvt_f32_f16<<<(NX / 4 + 255) / 256, 256, 0, stream>>>(k, hk, NX / 4);
  cvt_f32_f16<<<(NX / 4 + 255) / 256, 256, 0, stream>>>(v, hv, NX / 4);
  cvt_f32_f16<<<(NW / 4 + 255) / 256, 256, 0, stream>>>(Wq, hWq, NW / 4);
  cvt_f32_f16<<<(NW / 4 + 255) / 256, 256, 0, stream>>>(Wk, hWk, NW / 4);
  cvt_f32_f16<<<(NW / 4 + 255) / 256, 256, 0, stream>>>(Wv, hWv, NW / 4);
  cvt_f32_f16<<<(NW / 4 + 255) / 256, 256, 0, stream>>>(Wo, hWo, NW / 4);

  dim3 ggrid(Bn * Sn / 128, Dm / 64);
  gemm_xwt<false><<<ggrid, 128, 0, stream>>>(hq, hWq, bq, hQ, Bn * Sn, Dm, Dm);
  gemm_xwt<false><<<ggrid, 128, 0, stream>>>(hk, hWk, bk, hK, Bn * Sn, Dm, Dm);
  gemm_xwt<false><<<ggrid, 128, 0, stream>>>(hv, hWv, bv, hV, Bn * Sn, Dm, Dm);

  attn_kernel<<<dim3(Sn / 64, Hn, Bn), 128, 0, stream>>>(hQ, hK, hV, hA);

  gemm_xwt<true><<<ggrid, 128, 0, stream>>>(hA, hWo, bo, (float*)d_out,
                                            Bn * Sn, Dm, Dm);
}
